// RNNModel_46737834115352
// MI455X (gfx1250) — compile-verified, matrix operand order checked by reading, not run on hardware
//
#include <hip/hip_runtime.h>

// Problem dims (fixed by the reference)
constexpr int VSZ = 8192;   // vocab
constexpr int DSZ = 1024;   // hidden
constexpr int BSZ = 32;     // batch
constexpr int TSZ = 128;    // seq len

typedef __attribute__((ext_vector_type(16))) __bf16       v16bf;
typedef __attribute__((ext_vector_type(8)))  float        v8f;
typedef __attribute__((ext_vector_type(4)))  unsigned int u32x4;

union Frag {
    v16bf v;
    u32x4 q[2];
};

__device__ __forceinline__ unsigned short f2bf(float x) {
    unsigned u = __float_as_uint(x);
    u += 0x7FFFu + ((u >> 16) & 1u);   // round-to-nearest-even
    return (unsigned short)(u >> 16);
}

// Load one 16x32 bf16 fragment (A layout; also used for B as A-layout of B^T).
// Per ISA 7.12.2: lane L (group g=L>>4) holds row M=L&15,
// K runs [8g, 8g+8) and [16+8g, 16+8g+8)  -> two 16-byte loads.
__device__ __forceinline__ v16bf load_frag(const unsigned short* base, int stride, int lane) {
    Frag f;
    const int m  = lane & 15;
    const int kb = (lane >> 4) << 3;
    const unsigned short* p = base + m * stride + kb;
    f.q[0] = *(const u32x4*)(p);
    f.q[1] = *(const u32x4*)(p + 16);
    return f.v;
}

// Async global->LDS 16-byte copy (CDNA5 ASYNCcnt path).
// VDST = per-lane LDS byte address (low 32 bits of generic shared pointer),
// VADDR = 64-bit global address.
__device__ __forceinline__ void async_cp16(unsigned lds_byte_off, const void* gptr) {
    asm volatile("global_load_async_to_lds_b128 %0, %1, off"
                 :: "v"(lds_byte_off), "v"(gptr)
                 : "memory");
}

// ---------------------------------------------------------------------------
// Kernel 1: prep — bf16 weight copies, seed h0, zero grid-barrier counters
// ---------------------------------------------------------------------------
__global__ void __launch_bounds__(256)
prep_kernel(const float* __restrict__ W_hh, const float* __restrict__ W_ho,
            const float* __restrict__ h0,
            unsigned short* __restrict__ Whh_bf, unsigned short* __restrict__ Who_bf,
            unsigned short* __restrict__ Hbuf, unsigned* __restrict__ sync) {
    const long i0     = (long)blockIdx.x * blockDim.x + threadIdx.x;
    const long stride = (long)gridDim.x * blockDim.x;
    for (long i = i0; i < (long)VSZ * DSZ; i += stride) Who_bf[i] = f2bf(W_ho[i]);
    for (long i = i0; i < (long)DSZ * DSZ; i += stride) Whh_bf[i] = f2bf(W_hh[i]);
    for (long i = i0; i < (long)BSZ * DSZ; i += stride) Hbuf[i]   = f2bf(h0[i]);
    if (i0 < 8) sync[i0] = 0;
}

// ---------------------------------------------------------------------------
// Kernel 2: recurrence — persistent, 16 WGs, grid barrier per timestep.
// WG g owns hidden columns [64g, 64g+64). 8 waves = 2(M)x4(N) 16x16 frags.
// h_{t+1} = tanh( Wih[:,tok] + h_t @ Whh^T + b_ih + b_hh ), stored bf16.
// Operands (Whh_bf 2MB, h_t 64KB) are L2-resident; direct fragment loads.
// ---------------------------------------------------------------------------
__global__ void __launch_bounds__(256)
recurrence_kernel(const int* __restrict__ idx, const float* __restrict__ W_ih,
                  const float* __restrict__ b_ih, const float* __restrict__ b_hh,
                  const unsigned short* __restrict__ Whh_bf,
                  unsigned short* __restrict__ Hbuf, unsigned* __restrict__ sync) {
    __shared__ int s_tok[BSZ];
    const int tid  = threadIdx.x;
    const int lane = tid & 31;
    const int wave = tid >> 5;
    const int fm   = wave & 1;          // M (batch) fragment: rows 16*fm..
    const int fn   = wave >> 1;         // N (hidden) fragment: cols 16*fn..
    const int d0   = blockIdx.x * 64;

    const int n_lane = lane & 15;
    const int r_base = (lane >> 4) << 3;
    const int dcol   = d0 + fn * 16 + n_lane;
    const float bias_d = b_ih[dcol] + b_hh[dcol];

    volatile unsigned* g_cnt = sync;
    volatile unsigned* g_gen = sync + 1;

    for (int t = 0; t < TSZ; ++t) {
        if (tid < BSZ) s_tok[tid] = idx[tid * TSZ + t];
        __syncthreads();

        const unsigned short* Abase = Hbuf + (long)t * BSZ * DSZ + (long)(fm * 16) * DSZ;
        const unsigned short* Bbase = Whh_bf + (long)(d0 + fn * 16) * DSZ;

        v8f acc = {};
        for (int k = 0; k < DSZ; k += 32) {
            v16bf a = load_frag(Abase + k, DSZ, lane);
            v16bf b = load_frag(Bbase + k, DSZ, lane);
            acc = __builtin_amdgcn_wmma_f32_16x16x32_bf16(false, a, false, b,
                                                          (short)0, acc, false, false);
        }

        unsigned short* Hout = Hbuf + (long)(t + 1) * BSZ * DSZ;
        #pragma unroll
        for (int e = 0; e < 8; ++e) {
            const int brow = fm * 16 + r_base + e;
            const float xp = W_ih[(long)dcol * VSZ + s_tok[brow]];  // column gather
            const float h  = tanhf(acc[e] + xp + bias_d);
            Hout[brow * DSZ + dcol] = f2bf(h);
        }

        // ---- grid barrier (sense via generation counter) ----
        __syncthreads();
        if (tid == 0) {
            __threadfence();
            const unsigned g = *g_gen;
            const unsigned arrived = atomicAdd((unsigned*)g_cnt, 1u);
            if (arrived == gridDim.x - 1) {
                *g_cnt = 0;
                __threadfence();
                atomicAdd((unsigned*)g_gen, 1u);
            } else {
                while (*g_gen == g) __builtin_amdgcn_s_sleep(2);
            }
            __threadfence();
        }
        __syncthreads();
    }
}

// ---------------------------------------------------------------------------
// Kernel 3: logits GEMM — M=T*B=4096, N=V=8192, K=D=1024, bf16 WMMA, f32 acc.
// 128x128 WG tile, double-buffered LDS filled by async-to-LDS copies
// (ASYNCcnt pipeline), 8 waves each computing a 32x64 sub-tile.
// ---------------------------------------------------------------------------
constexpr int LDSP = 40;                       // padded LDS row stride (bf16)
constexpr unsigned BUFB = 128 * LDSP * 2;      // bytes per LDS buffer (10240)

__global__ void __launch_bounds__(256)
logits_kernel(const unsigned short* __restrict__ Abf,
              const unsigned short* __restrict__ Who_bf,
              const float* __restrict__ b_o, float* __restrict__ out) {
    __shared__ __align__(16) unsigned short sA[2][128 * LDSP];
    __shared__ __align__(16) unsigned short sB[2][128 * LDSP];

    const int tid  = threadIdx.x;
    const int lane = tid & 31;
    const int wave = tid >> 5;
    const int wm   = wave & 3;     // M offset 32*wm
    const int wn   = wave >> 2;    // N offset 64*wn
    const int gm   = blockIdx.y;   // 0..31  (M tiles of 128)
    const int gn   = blockIdx.x;   // 0..63  (N tiles of 128)

    const unsigned short* Ag = Abf    + (long)gm * 128 * DSZ;
    const unsigned short* Bg = Who_bf + (long)gn * 128 * DSZ;

    // Per-thread copy slots: 512 16B chunks per 128x32 tile; this thread owns
    // rows rr and rr+64 at column byte offset cc*2 (all 16B-aligned in LDS).
    const int rr = tid >> 2;
    const int cc = (tid & 3) << 3;
    const unsigned short* gA0 = Ag + (long)rr * DSZ + cc;
    const unsigned short* gA1 = gA0 + (long)64 * DSZ;
    const unsigned short* gB0 = Bg + (long)rr * DSZ + cc;
    const unsigned short* gB1 = gB0 + (long)64 * DSZ;
    const unsigned ldsA0 = (unsigned)(uintptr_t)&sA[0][rr * LDSP + cc];
    const unsigned ldsA1 = ldsA0 + (unsigned)(64 * LDSP * 2);
    const unsigned ldsB0 = (unsigned)(uintptr_t)&sB[0][rr * LDSP + cc];
    const unsigned ldsB1 = ldsB0 + (unsigned)(64 * LDSP * 2);

    v8f acc[2][4] = {};

    // Prologue: stage 0 into buffer 0 (4 async b128 per wave -> ASYNCcnt=4).
    async_cp16(ldsA0, gA0);
    async_cp16(ldsA1, gA1);
    async_cp16(ldsB0, gB0);
    async_cp16(ldsB1, gB1);

    for (int s = 0; s < 32; ++s) {
        const int buf = s & 1;
        __syncthreads();   // all waves finished reading the buffer stage s+1 overwrites
        if (s + 1 < 32) {
            const unsigned o = ((unsigned)((s + 1) & 1)) * BUFB;
            const int kadv = (s + 1) * 32;
            async_cp16(ldsA0 + o, gA0 + kadv);
            async_cp16(ldsA1 + o, gA1 + kadv);
            async_cp16(ldsB0 + o, gB0 + kadv);
            async_cp16(ldsB1 + o, gB1 + kadv);
            // Async loads retire in order: <=4 outstanding => stage s landed.
            asm volatile("s_wait_asynccnt 0x4" ::: "memory");
        } else {
            asm volatile("s_wait_asynccnt 0x0" ::: "memory");
        }
        __syncthreads();   // stage-s tiles visible to all waves

        const unsigned short* bufA = sA[buf];
        const unsigned short* bufB = sB[buf];
        const v16bf a0 = load_frag(bufA + (wm * 32 +  0) * LDSP, LDSP, lane);
        const v16bf a1 = load_frag(bufA + (wm * 32 + 16) * LDSP, LDSP, lane);
        #pragma unroll
        for (int j = 0; j < 4; ++j) {
            const v16bf b = load_frag(bufB + (wn * 64 + j * 16) * LDSP, LDSP, lane);
            acc[0][j] = __builtin_amdgcn_wmma_f32_16x16x32_bf16(false, a0, false, b,
                                                                (short)0, acc[0][j], false, false);
            acc[1][j] = __builtin_amdgcn_wmma_f32_16x16x32_bf16(false, a1, false, b,
                                                                (short)0, acc[1][j], false, false);
        }
    }

    const int n_lane = lane & 15;
    const int r_base = (lane >> 4) << 3;
    #pragma unroll
    for (int j = 0; j < 4; ++j) {
        const int col = gn * 128 + wn * 64 + j * 16 + n_lane;
        const float bo = b_o[col];
        #pragma unroll
        for (int i = 0; i < 2; ++i) {
            #pragma unroll
            for (int e = 0; e < 8; ++e) {
                const int row = gm * 128 + wm * 32 + i * 16 + r_base + e;
                out[(long)row * VSZ + col] = acc[i][j][e] + bo;
            }
        }
    }
}

// ---------------------------------------------------------------------------
// Kernel 4: in-place log_softmax over V=8192 per row; one WG per row,
// 32 values per thread kept in registers, two LDS tree reductions.
// ---------------------------------------------------------------------------
__global__ void __launch_bounds__(256)
logsoftmax_kernel(float* __restrict__ out) {
    __shared__ float red[256];
    const int  tid = threadIdx.x;
    const long row = blockIdx.x;
    float* p = out + row * VSZ;

    float xs[32];
    float m = -3.0e38f;
    #pragma unroll
    for (int j = 0; j < 32; ++j) {
        xs[j] = p[tid + 256 * j];
        m = fmaxf(m, xs[j]);
    }
    red[tid] = m;
    __syncthreads();
    for (int s = 128; s > 0; s >>= 1) {
        if (tid < s) red[tid] = fmaxf(red[tid], red[tid + s]);
        __syncthreads();
    }
    m = red[0];
    __syncthreads();

    float sum = 0.f;
    #pragma unroll
    for (int j = 0; j < 32; ++j) sum += expf(xs[j] - m);
    red[tid] = sum;
    __syncthreads();
    for (int s = 128; s > 0; s >>= 1) {
        if (tid < s) red[tid] += red[tid + s];
        __syncthreads();
    }
    const float lse = m + logf(red[0]);

    #pragma unroll
    for (int j = 0; j < 32; ++j) p[tid + 256 * j] = xs[j] - lse;
}

// ---------------------------------------------------------------------------
// Launch
// ---------------------------------------------------------------------------
extern "C" void kernel_launch(void* const* d_in, const int* in_sizes, int n_in,
                              void* d_out, int out_size, void* d_ws, size_t ws_size,
                              hipStream_t stream) {
    const int*   idx  = (const int*)  d_in[0];
    const float* h0   = (const float*)d_in[1];
    const float* W_ih = (const float*)d_in[2];
    const float* b_ih = (const float*)d_in[3];
    const float* W_hh = (const float*)d_in[4];
    const float* b_hh = (const float*)d_in[5];
    const float* W_ho = (const float*)d_in[6];
    const float* b_o  = (const float*)d_in[7];
    float* out = (float*)d_out;

    // Workspace layout (~27.3 MB):
    //   [0,256)                      : grid-barrier counters
    //   Hbuf : (T+1)*B*D bf16        : h0 + all h_t
    //   Whh_bf : D*D bf16
    //   Who_bf : V*D bf16
    char* ws = (char*)d_ws;
    unsigned*       sync   = (unsigned*)ws;
    unsigned short* Hbuf   = (unsigned short*)(ws + 256);
    unsigned short* Whh_bf = Hbuf + (long)(TSZ + 1) * BSZ * DSZ;
    unsigned short* Who_bf = Whh_bf + (long)DSZ * DSZ;

    prep_kernel<<<2048, 256, 0, stream>>>(W_hh, W_ho, h0, Whh_bf, Who_bf, Hbuf, sync);
    recurrence_kernel<<<16, 256, 0, stream>>>(idx, W_ih, b_ih, b_hh, Whh_bf, Hbuf, sync);
    logits_kernel<<<dim3(64, 32), 256, 0, stream>>>(Hbuf + (long)BSZ * DSZ, Who_bf, b_o, out);
    logsoftmax_kernel<<<4096, 256, 0, stream>>>(out);
}
